// GGNN_48704929137215
// MI455X (gfx1250) — matrix operand order repaired and checked by reading, not compiled, not run on hardware
//
#include <hip/hip_runtime.h>
#include <hip/hip_bf16.h>

#define N 1024
#define E 4
#define D 64
#define ANN 16
#define STEPS 5
#define B 4
#define EN (E * N)   // 4096

typedef __attribute__((ext_vector_type(16))) _Float16 v16h;
typedef __attribute__((ext_vector_type(8)))  float    v8f;
typedef int v4i_ __attribute__((vector_size(16)));

union F16x16 { v16h v; _Float16 h[16]; uint4 q[2]; };
union F16x8  { _Float16 h[8];  uint4 q; };
union F16x4  { _Float16 h[4];  uint2 q; };

#define WMMA_F16(a, b, c) \
  __builtin_amdgcn_wmma_f32_16x16x32_f16(false, (a), false, (b), (short)0, (c), false, false)

// gfx1250 async global->LDS staging (ASYNCcnt path); guarded, with direct-load fallback
#if defined(__has_builtin)
#if __has_builtin(__builtin_amdgcn_global_load_async_to_lds_b128)
#define HAVE_ASYNC_LDS 1
#endif
#endif
#ifndef HAVE_ASYNC_LDS
#define HAVE_ASYNC_LDS 0
#endif

// ---- fragment loaders (layouts per CDNA5 ISA 7.12.2, wave32) ----

// A-fragment 16x32 f16 from row-major f16, lda in elements (multiple of 8)
__device__ __forceinline__ v16h load_a16(const _Float16* p, int lda, int lane) {
  int row = lane & 15, ko = (lane >> 4) * 8;
  const _Float16* q = p + row * lda + ko;
  F16x16 u;
  u.q[0] = *(const uint4*)(q);        // K = ko .. ko+7
  u.q[1] = *(const uint4*)(q + 16);   // K = ko+16 .. ko+23
  return u.v;
}

// A-fragment 16x32 built from row-major f32 (convert on load)
__device__ __forceinline__ v16h load_a32(const float* p, int lda, int lane) {
  int row = lane & 15, ko = (lane >> 4) * 8;
  const float* q = p + row * lda + ko;
  v16h a;
#pragma unroll
  for (int i = 0; i < 8; ++i) a[i] = (_Float16)q[i];
#pragma unroll
  for (int i = 0; i < 8; ++i) a[i + 8] = (_Float16)q[i + 16];
  return a;
}

// B-fragment 32x16 f16 from B^T stored row-major (row = output col, K = ko..ko+7 and ko+16..ko+23 per lane half)
__device__ __forceinline__ v16h load_bT(const _Float16* p, int ldbt, int lane) {
  const _Float16* q = p + (lane & 15) * ldbt + (lane >> 4) * 8;
  F16x16 u;
  u.q[0] = *(const uint4*)(q);
  u.q[1] = *(const uint4*)(q + 16);
  return u.v;
}

// ---- prep: fold concat weights, convert to f16 (transposed-friendly) ----
__global__ void k_prep(const float* __restrict__ in_W, const float* __restrict__ rW,
                       const float* __restrict__ zW, const float* __restrict__ hW,
                       const float* __restrict__ o1W,
                       _Float16* __restrict__ inW16,
                       _Float16* __restrict__ Wra, _Float16* __restrict__ Wrh,
                       _Float16* __restrict__ Wza, _Float16* __restrict__ Wzh,
                       _Float16* __restrict__ Wha, _Float16* __restrict__ Whh,
                       _Float16* __restrict__ Wo1) {
  int tid = blockIdx.x * blockDim.x + threadIdx.x;
  int nt = gridDim.x * blockDim.x;
  for (int i = tid; i < E * D * D; i += nt) inW16[i] = (_Float16)in_W[i];
  for (int i = tid; i < D * D; i += nt) {
    int f = i >> 6, k = i & 63;
    Wra[i] = (_Float16)(rW[f * 192 + k] + rW[f * 192 + 64 + k]);
    Wrh[i] = (_Float16)(rW[f * 192 + 128 + k]);
    Wza[i] = (_Float16)(zW[f * 192 + k] + zW[f * 192 + 64 + k]);
    Wzh[i] = (_Float16)(zW[f * 192 + 128 + k]);
    Wha[i] = (_Float16)(hW[f * 192 + k] + hW[f * 192 + 64 + k]);
    Whh[i] = (_Float16)(hW[f * 192 + 128 + k]);
  }
  for (int i = tid; i < D * 96; i += nt) {
    int f = i / 96, k = i - f * 96;
    Wo1[i] = (k < D + ANN) ? (_Float16)o1W[f * (D + ANN) + k] : (_Float16)0.f;
  }
}

__global__ void k_copy_h(const float* __restrict__ src, float* __restrict__ dst) {
  int i = blockIdx.x * blockDim.x + threadIdx.x;
  if (i < B * N * D) dst[i] = src[i];
}

// A[:, :, :N*E] (row stride 2*N*E) -> f16 A16 (row stride N*E), 4 elems/thread
__global__ void k_cvtA(const float* __restrict__ A, _Float16* __restrict__ A16) {
  int g = blockIdx.x * blockDim.x + threadIdx.x;          // B*N*1024 groups
  if (g >= B * N * (EN / 4)) return;
  int row = g >> 10;
  int mi = (g & 1023) * 4;
  const float4 f = *(const float4*)(A + (size_t)row * (2 * EN) + mi);
  F16x4 u;
  u.h[0] = (_Float16)f.x; u.h[1] = (_Float16)f.y;
  u.h[2] = (_Float16)f.z; u.h[3] = (_Float16)f.w;
  *(uint2*)(A16 + (size_t)row * EN + mi) = u.q;
}

// ---- step kernel 1: in_states^T = (h @ in_W[e]^T + in_b[e])^T, f16 ----
__global__ __launch_bounds__(32) void k_instates(const float* __restrict__ hbuf,
                                                 const _Float16* __restrict__ inW16,
                                                 const float* __restrict__ in_b,
                                                 _Float16* __restrict__ inStT) {
  int lane = threadIdx.x;
  int blk = blockIdx.x;                 // b*E*(N/16)
  int n0 = (blk & 63) * 16;
  int e = (blk >> 6) & 3;
  int b = blk >> 8;
  const float* hrow = hbuf + (size_t)b * N * D + (size_t)n0 * D;
  v8f acc[4] = {};
#pragma unroll
  for (int kc = 0; kc < 2; ++kc) {
    v16h aF = load_a32(hrow + kc * 32, D, lane);
#pragma unroll
    for (int j = 0; j < 4; ++j) {
      v16h bF = load_bT(inW16 + e * D * D + (j * 16) * D + kc * 32, D, lane);
      acc[j] = WMMA_F16(aF, bF, acc[j]);
    }
  }
  int col = lane & 15, mb = (lane >> 4) * 8;
#pragma unroll
  for (int j = 0; j < 4; ++j) {
    int f = j * 16 + col;
    float bias = in_b[e * D + f];
    F16x8 o;
#pragma unroll
    for (int v = 0; v < 8; ++v) o.h[v] = (_Float16)(acc[j][v] + bias);
    // transposed store: inStT[b][f][e*N + n] — contiguous in n (8 halves = b128)
    *(uint4*)(inStT + (size_t)b * D * EN + (size_t)f * EN + e * N + n0 + mb) = o.q;
  }
}

// ---- step kernel 2: a_in = A16 @ in_states  (1024x4096 @ 4096x64 per batch) ----
// 4 waves per block K-split the 4096-deep reduction; fixed-order LDS reduce.
__global__ __launch_bounds__(128) void k_ain(const _Float16* __restrict__ A16,
                                             const _Float16* __restrict__ inStT,
                                             float* __restrict__ a_in) {
  __shared__ float red[4][16 * D];
#if HAVE_ASYNC_LDS
  __shared__ _Float16 stage[4][2][16 * 32];   // per-wave double buffer, 1KB each
#endif
  int tid = threadIdx.x;
  int lane = tid & 31;
  int w = tid >> 5;
  int n0 = ((int)blockIdx.x & 63) * 16;
  int b = (int)blockIdx.x >> 6;
  const int KCW = (EN / 32) / 4;              // 32 K-chunks per wave
  const _Float16* Ab = A16 + (size_t)b * N * EN + (size_t)n0 * EN;
  const _Float16* Bt = inStT + (size_t)b * D * EN;
  int kc0 = w * KCW;
  v8f acc[4] = {};

#if HAVE_ASYNC_LDS
  int srow = lane & 15, sseg = lane >> 4;
  auto issue = [&](int kc, int buf) {
#pragma unroll
    for (int i = 0; i < 2; ++i) {
      int seg = sseg + 2 * i;                 // 4 x 16B segments per 16x32-half row tile
      const _Float16* g = Ab + (size_t)srow * EN + kc * 32 + seg * 8;
      _Float16* l = &stage[w][buf][srow * 32 + seg * 8];
      __builtin_amdgcn_global_load_async_to_lds_b128(
          (__attribute__((address_space(1))) v4i_*)(_Float16*)g,
          (__attribute__((address_space(3))) v4i_*)l, 0, 0);
    }
  };
  issue(kc0, 0);
  for (int i = 0; i < KCW; ++i) {
    int kc = kc0 + i;
    if (i + 1 < KCW) {
      issue(kc + 1, (i + 1) & 1);
      asm volatile("s_wait_asynccnt 0x2" ::: "memory");   // chunk i complete (in-order)
    } else {
      asm volatile("s_wait_asynccnt 0x0" ::: "memory");
    }
    v16h aF = load_a16(&stage[w][i & 1][0], 32, lane);    // ds_load_b128 fragments
#pragma unroll
    for (int j = 0; j < 4; ++j) {
      v16h bF = load_bT(Bt + (size_t)(j * 16) * EN + kc * 32, EN, lane);
      acc[j] = WMMA_F16(aF, bF, acc[j]);
    }
  }
#else
  for (int i = 0; i < KCW; ++i) {
    int kc = kc0 + i;
    __builtin_prefetch(Ab + (lane & 15) * EN + kc * 32 + 256, 0, 1);
    v16h aF = load_a16(Ab + kc * 32, EN, lane);
#pragma unroll
    for (int j = 0; j < 4; ++j) {
      v16h bF = load_bT(Bt + (size_t)(j * 16) * EN + kc * 32, EN, lane);
      acc[j] = WMMA_F16(aF, bF, acc[j]);
    }
  }
#endif

  int col = lane & 15, mb = (lane >> 4) * 8;
#pragma unroll
  for (int j = 0; j < 4; ++j)
#pragma unroll
    for (int v = 0; v < 8; ++v) red[w][(mb + v) * D + j * 16 + col] = acc[j][v];
  __syncthreads();
  float* o = a_in + (size_t)b * N * D + (size_t)n0 * D;   // strip is row-major 16xD == linear
  for (int i = tid; i < 16 * D; i += 128)
    o[i] = red[0][i] + red[1][i] + red[2][i] + red[3][i];
}

// ---- step kernel 3: fused GRU gates + state update (in place on h) ----
__global__ __launch_bounds__(32) void k_gate(const float* __restrict__ a_in,
                                             float* __restrict__ hbuf,
                                             const _Float16* __restrict__ Wra,
                                             const _Float16* __restrict__ Wrh,
                                             const _Float16* __restrict__ Wza,
                                             const _Float16* __restrict__ Wzh,
                                             const _Float16* __restrict__ Wha,
                                             const _Float16* __restrict__ Whh,
                                             const float* __restrict__ rb,
                                             const float* __restrict__ zb,
                                             const float* __restrict__ hbias) {
  __shared__ _Float16 rhTile[16 * D];
  int lane = threadIdx.x;
  int n0 = (blockIdx.x & 63) * 16;
  int b = blockIdx.x >> 6;
  const float* arow = a_in + (size_t)b * N * D + (size_t)n0 * D;
  float* hrow = hbuf + (size_t)b * N * D + (size_t)n0 * D;
  v8f accR[4] = {}, accZ[4] = {}, accH[4] = {};
#pragma unroll
  for (int kc = 0; kc < 2; ++kc) {
    v16h aF = load_a32(arow + kc * 32, D, lane);
    v16h hF = load_a32(hrow + kc * 32, D, lane);
#pragma unroll
    for (int j = 0; j < 4; ++j) {
      accR[j] = WMMA_F16(aF, load_bT(Wra + (j * 16) * D + kc * 32, D, lane), accR[j]);
      accR[j] = WMMA_F16(hF, load_bT(Wrh + (j * 16) * D + kc * 32, D, lane), accR[j]);
      accZ[j] = WMMA_F16(aF, load_bT(Wza + (j * 16) * D + kc * 32, D, lane), accZ[j]);
      accZ[j] = WMMA_F16(hF, load_bT(Wzh + (j * 16) * D + kc * 32, D, lane), accZ[j]);
      accH[j] = WMMA_F16(aF, load_bT(Wha + (j * 16) * D + kc * 32, D, lane), accH[j]);
    }
  }
  int col = lane & 15, mb = (lane >> 4) * 8;
  float hv[4][8], zv[4][8];
#pragma unroll
  for (int j = 0; j < 4; ++j) {
    int f = j * 16 + col;
    float rbv = rb[f], zbv = zb[f];
#pragma unroll
    for (int v = 0; v < 8; ++v) {
      float hval = hrow[(mb + v) * D + f];
      hv[j][v] = hval;
      float r = 1.f / (1.f + __expf(-(accR[j][v] + rbv)));
      zv[j][v] = 1.f / (1.f + __expf(-(accZ[j][v] + zbv)));
      rhTile[(mb + v) * D + f] = (_Float16)(r * hval);   // C-layout -> row-major LDS
    }
  }
  __syncthreads();
#pragma unroll
  for (int kc = 0; kc < 2; ++kc) {
    v16h rhF = load_a16(&rhTile[kc * 32], D, lane);      // ds_load fragments
#pragma unroll
    for (int j = 0; j < 4; ++j)
      accH[j] = WMMA_F16(rhF, load_bT(Whh + (j * 16) * D + kc * 32, D, lane), accH[j]);
  }
#pragma unroll
  for (int j = 0; j < 4; ++j) {
    int f = j * 16 + col;
    float hbv = hbias[f];
#pragma unroll
    for (int v = 0; v < 8; ++v) {
      float hh = tanhf(accH[j][v] + hbv);
      float z = zv[j][v];
      hrow[(mb + v) * D + f] = (1.f - z) * hv[j][v] + z * hh;
    }
  }
}

// ---- readout: out[b,n] = tanh([h, ann] @ o1W^T + o1b) . o2W + o2b ----
__global__ __launch_bounds__(32) void k_out(const float* __restrict__ hbuf,
                                            const float* __restrict__ ann,
                                            const _Float16* __restrict__ Wo1,
                                            const float* __restrict__ o1b,
                                            const float* __restrict__ o2W,
                                            const float* __restrict__ o2b,
                                            float* __restrict__ out) {
  __shared__ float t[16 * D];
  int lane = threadIdx.x;
  int n0 = (blockIdx.x & 63) * 16;
  int b = blockIdx.x >> 6;
  const float* hrow = hbuf + (size_t)b * N * D + (size_t)n0 * D;
  const float* arow = ann + (size_t)b * N * ANN + (size_t)n0 * ANN;
  v8f acc[4] = {};
#pragma unroll
  for (int kc = 0; kc < 2; ++kc) {
    v16h aF = load_a32(hrow + kc * 32, D, lane);
#pragma unroll
    for (int j = 0; j < 4; ++j)
      acc[j] = WMMA_F16(aF, load_bT(Wo1 + (j * 16) * 96 + kc * 32, 96, lane), acc[j]);
  }
  {  // K-chunk 2: annotation cols (K=64..79 valid, 80..95 zero-padded)
    int row = lane & 15, ko = (lane >> 4) * 8;
    v16h aF;
#pragma unroll
    for (int i = 0; i < 8; ++i) aF[i] = (_Float16)arow[row * ANN + ko + i];
#pragma unroll
    for (int i = 0; i < 8; ++i) aF[i + 8] = (_Float16)0.f;
#pragma unroll
    for (int j = 0; j < 4; ++j)
      acc[j] = WMMA_F16(aF, load_bT(Wo1 + (j * 16) * 96 + 64, 96, lane), acc[j]);
  }
  int col = lane & 15, mb = (lane >> 4) * 8;
#pragma unroll
  for (int j = 0; j < 4; ++j) {
    float bb = o1b[j * 16 + col];
#pragma unroll
    for (int v = 0; v < 8; ++v) t[(mb + v) * D + j * 16 + col] = tanhf(acc[j][v] + bb);
  }
  __syncthreads();
  if (lane < 16) {
    float s = o2b[0];
#pragma unroll
    for (int f = 0; f < D; ++f) s += t[lane * D + f] * o2W[f];
    out[(size_t)b * N + n0 + lane] = s;
  }
}

extern "C" void kernel_launch(void* const* d_in, const int* in_sizes, int n_in,
                              void* d_out, int out_size, void* d_ws, size_t ws_size,
                              hipStream_t stream) {
  const float* prop = (const float*)d_in[0];
  const float* ann  = (const float*)d_in[1];
  const float* A    = (const float*)d_in[2];
  const float* in_W = (const float*)d_in[3];
  const float* in_b = (const float*)d_in[4];
  // d_in[5]=out_W, d_in[6]=out_b are unused by the reference (a_out = a_in)
  const float* rW   = (const float*)d_in[7];
  const float* rb   = (const float*)d_in[8];
  const float* zW   = (const float*)d_in[9];
  const float* zb   = (const float*)d_in[10];
  const float* hW   = (const float*)d_in[11];
  const float* hb   = (const float*)d_in[12];
  const float* o1W  = (const float*)d_in[13];
  const float* o1b  = (const float*)d_in[14];
  const float* o2W  = (const float*)d_in[15];
  const float* o2b  = (const float*)d_in[16];
  float* out = (float*)d_out;

  char* ws = (char*)d_ws;
  size_t off = 0;
  auto alloc = [&](size_t bytes) { char* p = ws + off; off = (off + bytes + 255) & ~(size_t)255; return p; };
  _Float16* A16    = (_Float16*)alloc((size_t)B * N * EN * 2);   // 32 MB
  float*    hbuf   = (float*)alloc((size_t)B * N * D * 4);       // 1 MB
  float*    a_in   = (float*)alloc((size_t)B * N * D * 4);       // 1 MB
  _Float16* inStT  = (_Float16*)alloc((size_t)B * D * EN * 2);   // 2 MB
  _Float16* inW16  = (_Float16*)alloc((size_t)E * D * D * 2);
  _Float16* Wra = (_Float16*)alloc(D * D * 2);
  _Float16* Wrh = (_Float16*)alloc(D * D * 2);
  _Float16* Wza = (_Float16*)alloc(D * D * 2);
  _Float16* Wzh = (_Float16*)alloc(D * D * 2);
  _Float16* Wha = (_Float16*)alloc(D * D * 2);
  _Float16* Whh = (_Float16*)alloc(D * D * 2);
  _Float16* Wo1 = (_Float16*)alloc(D * 96 * 2);
  (void)ws_size; (void)in_sizes; (void)n_in; (void)out_size;

  k_prep<<<64, 256, 0, stream>>>(in_W, rW, zW, hW, o1W, inW16, Wra, Wrh, Wza, Wzh, Wha, Whh, Wo1);
  k_copy_h<<<(B * N * D) / 256, 256, 0, stream>>>(prop, hbuf);
  k_cvtA<<<(B * N * (EN / 4)) / 256, 256, 0, stream>>>(A, A16);

  for (int step = 0; step < STEPS; ++step) {
    k_instates<<<B * E * (N / 16), 32, 0, stream>>>(hbuf, inW16, in_b, inStT);
    k_ain<<<B * (N / 16), 128, 0, stream>>>(A16, inStT, a_in);
    k_gate<<<B * (N / 16), 32, 0, stream>>>(a_in, hbuf, Wra, Wrh, Wza, Wzh, Wha, Whh, rb, zb, hb);
  }
  k_out<<<B * (N / 16), 32, 0, stream>>>(hbuf, ann, Wo1, o1b, o2W, o2b, out);
}
